// DPVGAE_OGB_41351945126001
// MI455X (gfx1250) — matrix operand order repaired and checked by reading, not compiled, not run on hardware
//
#include <hip/hip_runtime.h>
#include <hip/hip_bf16.h>
#include <math.h>

// ---------------------------------------------------------------------------
// Problem constants (match reference)
// ---------------------------------------------------------------------------
#define NN    8192
#define FDIM  128
#define EE    262144
#define ENCD  256
#define HIDD  128
#define DECD  64
#define KC    10

typedef _Float16 half8 __attribute__((ext_vector_type(8)));
typedef _Float16 v16h  __attribute__((ext_vector_type(16)));
typedef float    v8f   __attribute__((ext_vector_type(8)));
typedef float    fl4   __attribute__((ext_vector_type(4)));

// ---------------------------------------------------------------------------
// Helpers
// ---------------------------------------------------------------------------
__device__ __forceinline__ float wave_sum(float v) {
  #pragma unroll
  for (int m = 16; m >= 1; m >>= 1) v += __shfl_xor(v, m, 32);
  return v;
}

__device__ __forceinline__ unsigned f2u_order(float f) {
  unsigned u = __float_as_uint(f);
  return (u & 0x80000000u) ? ~u : (u | 0x80000000u);
}

__device__ __forceinline__ float log_sigmoid(float x) {
  return (x >= 0.f) ? -log1pf(__expf(-x)) : (x - log1pf(__expf(x)));
}

// CDNA5 async global->LDS copy (ASYNCcnt tracked). The LDS destination byte
// address travels in a VGPR (low 32 bits of the generic shared pointer).
__device__ __forceinline__ void async_copy_b128(unsigned lds_off,
                                                const _Float16* g) {
  asm volatile("global_load_async_to_lds_b128 %0, %1, off"
               :: "v"(lds_off), "v"(g) : "memory");
}
__device__ __forceinline__ void wait_async0() {
  asm volatile("s_wait_asynccnt 0" ::: "memory");
}
__device__ __forceinline__ unsigned lds_addr(const void* p) {
  return (unsigned)(unsigned long long)p;
}

// A tile loader: 16x32 f16, A row-major (lda halves). CDNA5 layout:
// lane<16: M=lane, K = kh8+0..7 (v0..3) and 16+kh8+0..7 (v4..7); kh8=(lane>>4)*8
__device__ __forceinline__ v16h load_A16(const _Float16* __restrict__ A, int lda,
                                         int row0, int k0, int lane) {
  const int m  = lane & 15;
  const int kh = (lane >> 4) * 8;
  const _Float16* p = A + (size_t)(row0 + m) * lda + k0 + kh;
  half8 lo = *reinterpret_cast<const half8*>(p);
  half8 hi = *reinterpret_cast<const half8*>(p + 16);
  v16h a;
  #pragma unroll
  for (int i = 0; i < 8; i++) { a[i] = lo[i]; a[8 + i] = hi[i]; }
  return a;
}

// B tile loader: 32x16 f16, B stored K-contiguous per column: Bt[n*ldb + k].
// lane holds column n=(lane&15); K = kh8+0..7 (v0..3) and 16+kh8+0..7 (v4..7); kh8=(lane>>4)*8
__device__ __forceinline__ v16h load_B16(const _Float16* __restrict__ Bt, int ldb,
                                         int n0, int k0, int lane) {
  const int n  = n0 + (lane & 15);
  const int kh = (lane >> 4) * 8;
  const _Float16* p = Bt + (size_t)n * ldb + k0 + kh;
  half8 lo = *reinterpret_cast<const half8*>(p);
  half8 hi = *reinterpret_cast<const half8*>(p + 16);
  v16h b;
  #pragma unroll
  for (int i = 0; i < 8; i++) { b[i] = lo[i]; b[8 + i] = hi[i]; }
  return b;
}

__device__ __forceinline__ v8f wmma16(v16h a, v16h b, v8f c) {
  return __builtin_amdgcn_wmma_f32_16x16x32_f16(false, a, false, b, (short)0, c,
                                                false, false);
}

// ---------------------------------------------------------------------------
// Utility kernels
// ---------------------------------------------------------------------------
__global__ void k_fill_f32(float* __restrict__ p, long long n, float v) {
  long long i = (long long)blockIdx.x * blockDim.x + threadIdx.x;
  long long st = (long long)gridDim.x * blockDim.x;
  for (; i < n; i += st) p[i] = v;
}

__global__ void k_zero_u32(unsigned* __restrict__ p, int n) {
  int i = blockIdx.x * blockDim.x + threadIdx.x;
  if (i < n) p[i] = 0u;
}

// transpose + f32->f16: src is (R x C) row-major, dst is (C x R): dst[c*R+r]
__global__ void k_transpose_h(const float* __restrict__ src, _Float16* __restrict__ d,
                              int R, int C) {
  int t = blockIdx.x * blockDim.x + threadIdx.x;
  if (t >= R * C) return;
  int r = t / C, c = t - r * C;
  d[(size_t)c * R + r] = (_Float16)src[t];
}

// ---------------------------------------------------------------------------
// x_ban = softmax(mask/alpha) @ x  -- ONE pass (online softmax, flash-style).
// 4 waves / block, each wave owns a 16-row tile; B (x^T, f16) panel staged to
// LDS with async double-buffering shared by all 4 waves.
// ---------------------------------------------------------------------------
__global__ void k_softmax_gemm(const float* __restrict__ mn,
                               const _Float16* __restrict__ xht,
                               _Float16* __restrict__ xbanh) {
  __shared__ __align__(16) _Float16 Bs[2][FDIM * 32];  // [n*32 + k]
  const int tid  = threadIdx.x;
  const int lane = tid & 31;
  const int wv   = tid >> 5;
  const int row0 = (blockIdx.x * 4 + wv) * 16;
  const int m    = lane & 15;
  const int kh8  = (lane >> 4) * 8;
  const int rb   = (lane >> 4) * 8;
  const float* Ap = mn + (size_t)(row0 + m) * NN;

  // stage chunk 0 (each thread owns one column n=tid, 32 K-values = 4x b128)
  {
    const _Float16* g = xht + (size_t)tid * NN;
    unsigned l = lds_addr(&Bs[0][tid * 32]);
    async_copy_b128(l,      g);
    async_copy_b128(l + 16, g + 8);
    async_copy_b128(l + 32, g + 16);
    async_copy_b128(l + 48, g + 24);
  }
  wait_async0();
  __syncthreads();

  v8f acc[8] = {};
  float runmax = -3.0e38f, runsum = 0.f;

  #pragma unroll 1
  for (int k0 = 0; k0 < NN; k0 += 32) {
    const int cur = (k0 >> 5) & 1;
    if (k0 + 32 < NN) {                       // async prefetch next panel
      const _Float16* g = xht + (size_t)tid * NN + (k0 + 32);
      unsigned l = lds_addr(&Bs[1 - cur][tid * 32]);
      async_copy_b128(l,      g);
      async_copy_b128(l + 16, g + 8);
      async_copy_b128(l + 32, g + 16);
      async_copy_b128(l + 48, g + 24);
    }
    __builtin_prefetch(Ap + k0 + 1024, 0, 0);
    fl4 q0 = *reinterpret_cast<const fl4*>(Ap + k0 + kh8);
    fl4 q1 = *reinterpret_cast<const fl4*>(Ap + k0 + kh8 + 4);
    fl4 r0 = *reinterpret_cast<const fl4*>(Ap + k0 + 16 + kh8);
    fl4 r1 = *reinterpret_cast<const fl4*>(Ap + k0 + 16 + kh8 + 4);

    // online softmax update (lane pair (lane, lane^16) shares row m)
    float vmax = fmaxf(fmaxf(fmaxf(q0[0], q0[1]), fmaxf(q0[2], q0[3])),
                       fmaxf(fmaxf(q1[0], q1[1]), fmaxf(q1[2], q1[3])));
    vmax = fmaxf(vmax, fmaxf(fmaxf(fmaxf(r0[0], r0[1]), fmaxf(r0[2], r0[3])),
                             fmaxf(fmaxf(r1[0], r1[1]), fmaxf(r1[2], r1[3]))));
    float pmax = fmaxf(vmax, __shfl_xor(vmax, 16, 32));
    float nmax = fmaxf(runmax, pmax);
    float factor = __expf((runmax - nmax) * 2.f);

    v16h a;
    float ls = 0.f;
    #pragma unroll
    for (int i = 0; i < 4; i++) {
      float e0 = __expf((q0[i] - nmax) * 2.f);
      float e1 = __expf((q1[i] - nmax) * 2.f);
      float e2 = __expf((r0[i] - nmax) * 2.f);
      float e3 = __expf((r1[i] - nmax) * 2.f);
      a[i] = (_Float16)e0; a[4 + i] = (_Float16)e1;
      a[8 + i] = (_Float16)e2; a[12 + i] = (_Float16)e3;
      ls += e0 + e1 + e2 + e3;
    }
    float ps = ls + __shfl_xor(ls, 16, 32);
    runsum = runsum * factor + ps;
    runmax = nmax;

    if (__any(factor != 1.0f)) {              // rescale accumulators (rare)
      #pragma unroll
      for (int i = 0; i < 8; i++) {
        float f = __shfl(factor, rb + i, 32);
        #pragma unroll
        for (int t = 0; t < 8; t++) acc[t][i] *= f;
      }
    }

    #pragma unroll
    for (int t = 0; t < 8; t++) {
      v16h b = load_B16(&Bs[cur][0], 32, t * 16, 0, lane);
      acc[t] = wmma16(a, b, acc[t]);
    }
    wait_async0();
    __syncthreads();
  }

  const int col = lane & 15;
  float inv = 1.f / runsum;                   // per lane: row m
  float invs[8];
  #pragma unroll
  for (int i = 0; i < 8; i++) invs[i] = __shfl(inv, rb + i, 32);
  #pragma unroll
  for (int t = 0; t < 8; t++)
    #pragma unroll
    for (int i = 0; i < 8; i++)
      xbanh[(size_t)(row0 + rb + i) * FDIM + t * 16 + col] =
          (_Float16)(acc[t][i] * invs[i]);
}

// ---------------------------------------------------------------------------
// Generic WMMA GEMM: C(MxNc) = A(MxK) @ B(KxNc); A row-major f16 (lda),
// Bt column-contiguous f16 (ldb=K). 4 waves/block share an async double-
// buffered LDS B panel. grid = (M/64, Nc/64), block = 128.
// ---------------------------------------------------------------------------
__global__ void k_gemm(const _Float16* __restrict__ A, int lda,
                       const _Float16* __restrict__ Bt, int ldb,
                       const float* __restrict__ bias,
                       float* __restrict__ Cf, _Float16* __restrict__ Ch,
                       int ldc, int Ktot, int relu) {
  __shared__ __align__(16) _Float16 Bs[2][64 * 32];    // [n*32 + k]
  const int tid  = threadIdx.x;
  const int lane = tid & 31;
  const int wv   = tid >> 5;
  const int row0 = (blockIdx.x * 4 + wv) * 16;
  const int n0   = blockIdx.y * 64;
  const int sn   = tid >> 1;            // staging column 0..63
  const int sk   = (tid & 1) * 16;      // staging K half
  {
    const _Float16* g = Bt + (size_t)(n0 + sn) * ldb + sk;
    unsigned l = lds_addr(&Bs[0][sn * 32 + sk]);
    async_copy_b128(l,      g);
    async_copy_b128(l + 16, g + 8);
  }
  wait_async0();
  __syncthreads();

  v8f acc[4] = {};
  for (int k0 = 0; k0 < Ktot; k0 += 32) {
    const int cur = (k0 >> 5) & 1;
    if (k0 + 32 < Ktot) {
      const _Float16* g = Bt + (size_t)(n0 + sn) * ldb + (k0 + 32) + sk;
      unsigned l = lds_addr(&Bs[1 - cur][sn * 32 + sk]);
      async_copy_b128(l,      g);
      async_copy_b128(l + 16, g + 8);
    }
    v16h a = load_A16(A, lda, row0, k0, lane);
    #pragma unroll
    for (int t = 0; t < 4; t++) {
      v16h b = load_B16(&Bs[cur][0], 32, t * 16, 0, lane);
      acc[t] = wmma16(a, b, acc[t]);
    }
    wait_async0();
    __syncthreads();
  }
  const int col = lane & 15;
  const int rb  = (lane >> 4) * 8;
  #pragma unroll
  for (int t = 0; t < 4; t++) {
    const int n = n0 + t * 16 + col;
    const float bv = bias ? bias[n] : 0.f;
    #pragma unroll
    for (int i = 0; i < 8; i++) {
      float v = acc[t][i] + bv;
      if (relu) v = fmaxf(v, 0.f);
      const size_t idx = (size_t)(row0 + rb + i) * ldc + n;
      if (Cf) Cf[idx] = v;
      if (Ch) Ch[idx] = (_Float16)v;
    }
  }
}

// ---------------------------------------------------------------------------
// Topology filter: norms, sim, radix top-k select, masks
// ---------------------------------------------------------------------------
__global__ void k_norms(const float* __restrict__ x, float* __restrict__ invn) {
  int i = blockIdx.x * blockDim.x + threadIdx.x;
  if (i >= NN) return;
  const float* p = x + (size_t)i * FDIM;
  float s = 0.f;
  for (int f = 0; f < FDIM; f += 4) {
    fl4 v = *reinterpret_cast<const fl4*>(p + f);
    s += v[0] * v[0] + v[1] * v[1] + v[2] * v[2] + v[3] * v[3];
  }
  invn[i] = 1.f / fmaxf(sqrtf(s), 1e-8f);
}

__global__ void k_sim(const float* __restrict__ x, const float* __restrict__ invn,
                      const int* __restrict__ src, const int* __restrict__ dst,
                      float* __restrict__ sim, unsigned* __restrict__ keys,
                      unsigned* __restrict__ ctrl) {
  int e = blockIdx.x * blockDim.x + threadIdx.x;
  if (e >= EE) return;
  const int s = src[e], d = dst[e];
  const float* ps = x + (size_t)s * FDIM;
  const float* pd = x + (size_t)d * FDIM;
  float dot = 0.f;
  for (int f = 0; f < FDIM; f += 4) {
    fl4 a = *reinterpret_cast<const fl4*>(ps + f);
    fl4 b = *reinterpret_cast<const fl4*>(pd + f);
    dot += a[0] * b[0] + a[1] * b[1] + a[2] * b[2] + a[3] * b[3];
  }
  float sv = dot * invn[s] * invn[d];
  sim[e] = sv;
  bool pruned = !(sv >= 0.5f);
  keys[e] = pruned ? f2u_order(sv) : 0u;
  if (pruned) atomicAdd(&ctrl[0], 1u);
}

// ctrl: [0]=pruned cnt [1]=k [2]=prefix [3]=remaining [5]=ticket [6]=wpur cnt [7]=threshold
__global__ void k_sel_init(unsigned* __restrict__ ctrl, int mode) {
  if (threadIdx.x != 0 || blockIdx.x != 0) return;
  unsigned k;
  if (mode == 0) k = (unsigned)floorf(0.7f * (1.0f - 1.0f / 200.0f) * (float)ctrl[0]);
  else           k = (unsigned)floorf(0.7f * (float)ctrl[6]);
  ctrl[1] = k; ctrl[2] = 0u; ctrl[3] = k; ctrl[5] = 0u;
}

__global__ void k_hist(const unsigned* __restrict__ keys, unsigned* __restrict__ ctrl,
                       unsigned* __restrict__ hist, unsigned mask, int shift) {
  int e = blockIdx.x * blockDim.x + threadIdx.x;
  if (e >= EE) return;
  unsigned want = ctrl[2] & mask;
  unsigned k = keys[e];
  if ((k & mask) == want) atomicAdd(&hist[(k >> shift) & 255u], 1u);
}

__global__ void k_sel_bin(unsigned* __restrict__ ctrl, unsigned* __restrict__ hist,
                          int shift) {
  if (threadIdx.x != 0 || blockIdx.x != 0) return;
  unsigned k = ctrl[1];
  if (k > 0u) {
    unsigned r = ctrl[3];
    int sel = -1;
    for (int b = 255; b >= 0; --b) {
      unsigned c = hist[b];
      if (c < r) r -= c; else { sel = b; break; }
    }
    if (sel >= 0) ctrl[2] |= ((unsigned)sel) << shift;
    ctrl[3] = r;
  }
  if (shift == 0) { ctrl[7] = (k > 0u) ? ctrl[2] : 0xFFFFFFFFu; ctrl[5] = 0u; }
  for (int b = 0; b < 256; b++) hist[b] = 0u;
}

__global__ void k_mark_pur(const float* __restrict__ sim, const unsigned* __restrict__ keys,
                           unsigned* __restrict__ ctrl, float* __restrict__ wpur) {
  int e = blockIdx.x * blockDim.x + threadIdx.x;
  if (e >= EE) return;
  bool kept = sim[e] >= 0.5f;
  bool sel = false;
  if (!kept && ctrl[1] > 0u) {
    unsigned k = keys[e], T = ctrl[7];
    if (k > T) sel = true;
    else if (k == T) { unsigned t = atomicAdd(&ctrl[5], 1u); sel = (t < ctrl[3]); }
  }
  float wv = (kept || sel) ? 1.f : 0.f;
  wpur[e] = wv;
  if (wv > 0.f) atomicAdd(&ctrl[6], 1u);
}

__global__ void k_keys_v2(const float* __restrict__ perm, const float* __restrict__ wpur,
                          unsigned* __restrict__ keys) {
  int e = blockIdx.x * blockDim.x + threadIdx.x;
  if (e >= EE) return;
  keys[e] = (wpur[e] > 0.f) ? f2u_order(perm[e]) : 0u;
}

__global__ void k_mark_v2(const unsigned* __restrict__ keys, const float* __restrict__ wpur,
                          unsigned* __restrict__ ctrl, float* __restrict__ wv2) {
  int e = blockIdx.x * blockDim.x + threadIdx.x;
  if (e >= EE) return;
  bool sel = false;
  if (wpur[e] > 0.f && ctrl[1] > 0u) {
    unsigned k = keys[e], T = ctrl[7];
    if (k > T) sel = true;
    else if (k == T) { unsigned t = atomicAdd(&ctrl[5], 1u); sel = (t < ctrl[3]); }
  }
  wv2[e] = sel ? 1.f : 0.f;
}

// ---------------------------------------------------------------------------
// Graph aggregation
// ---------------------------------------------------------------------------
__global__ void k_deg_add(const int* __restrict__ idx, const float* __restrict__ w,
                          float* __restrict__ deg) {
  int e = blockIdx.x * blockDim.x + threadIdx.x;
  if (e >= EE) return;
  float we = w ? w[e] : 1.f;
  if (we != 0.f) atomicAdd(&deg[idx[e]], we);
}

// GCN (deg != null): agg[dst] += hl[src] * rsqrt(deg[s]) * rsqrt(deg[d]) * w
// GIN (deg == null): agg[dst] += hl[src] * w
__global__ void k_scatter(const float* __restrict__ hl, int F,
                          const int* __restrict__ src, const int* __restrict__ dst,
                          const float* __restrict__ w, const float* __restrict__ deg,
                          float* __restrict__ agg) {
  long long tid = (long long)blockIdx.x * blockDim.x + threadIdx.x;
  int e = (int)(tid >> 5);
  int lane = (int)(tid & 31);
  if (e >= EE) return;
  float we = w ? w[e] : 1.f;
  if (we == 0.f) return;
  const int s = src[e], d = dst[e];
  float coef = we;
  if (deg) coef *= rsqrtf(deg[s]) * rsqrtf(deg[d]);
  const float* hs = hl + (size_t)s * F;
  float* ad = agg + (size_t)d * F;
  for (int f = lane; f < F; f += 32) atomicAdd(&ad[f], hs[f] * coef);
}

__global__ void k_gcn_combine(const float* __restrict__ agg, const float* __restrict__ hl,
                              const float* __restrict__ deg, const float* __restrict__ bias,
                              int F, int relu, float* __restrict__ outf,
                              _Float16* __restrict__ outh) {
  long long t = (long long)blockIdx.x * blockDim.x + threadIdx.x;
  if (t >= (long long)NN * F) return;
  int i = (int)(t / F), f = (int)(t - (long long)i * F);
  float v = agg[t] + hl[t] * (1.f / deg[i]) + bias[f];
  if (relu) v = fmaxf(v, 0.f);
  if (outf) outf[t] = v;
  if (outh) outh[t] = (_Float16)v;
}

__global__ void k_gin_combine(const float* __restrict__ agg, const float* __restrict__ h,
                              _Float16* __restrict__ outh, long long n) {
  long long t = (long long)blockIdx.x * blockDim.x + threadIdx.x;
  if (t >= n) return;
  outh[t] = (_Float16)(agg[t] + h[t]);
}

// ---------------------------------------------------------------------------
// Losses
// ---------------------------------------------------------------------------
__global__ void k_linkloss(const float* __restrict__ z, const int* __restrict__ idx,
                           float flip, float* __restrict__ acc) {
  int e = blockIdx.x * blockDim.x + threadIdx.x;
  float v = 0.f;
  if (e < EE) {
    const int s = idx[e], d = idx[EE + e];
    const float* ps = z + (size_t)s * HIDD;
    const float* pd = z + (size_t)d * HIDD;
    float dot = 0.f;
    for (int f = 0; f < HIDD; f += 4) {
      fl4 a = *reinterpret_cast<const fl4*>(ps + f);
      fl4 b = *reinterpret_cast<const fl4*>(pd + f);
      dot += a[0] * b[0] + a[1] * b[1] + a[2] * b[2] + a[3] * b[3];
    }
    v = log_sigmoid(flip * dot);
  }
  v = wave_sum(v);
  if ((threadIdx.x & 31) == 0) atomicAdd(acc, v);
}

__global__ void k_mse(const float* __restrict__ a, const float* __restrict__ b,
                      long long n, float* __restrict__ acc) {
  long long i = (long long)blockIdx.x * blockDim.x + threadIdx.x;
  long long st = (long long)gridDim.x * blockDim.x;
  float s = 0.f;
  for (; i < n; i += st) { float d = a[i] - b[i]; s += d * d; }
  s = wave_sum(s);
  if ((threadIdx.x & 31) == 0) atomicAdd(acc, s);
}

__global__ void k_fg(const float* __restrict__ h1, const float* __restrict__ h2,
                     float* __restrict__ acc) {
  int i = blockIdx.x * blockDim.x + threadIdx.x;
  if (i >= NN) return;
  const float* a = h1 + (size_t)i * DECD;
  const float* b = h2 + (size_t)i * DECD;
  float na = 0.f, nb = 0.f, d = 0.f;
  for (int f = 0; f < DECD; f += 4) {
    fl4 u = *reinterpret_cast<const fl4*>(a + f);
    fl4 v = *reinterpret_cast<const fl4*>(b + f);
    na += u[0]*u[0]+u[1]*u[1]+u[2]*u[2]+u[3]*u[3];
    nb += v[0]*v[0]+v[1]*v[1]+v[2]*v[2]+v[3]*v[3];
    d  += u[0]*v[0]+u[1]*v[1]+u[2]*v[2]+u[3]*v[3];
  }
  float s = d / (fmaxf(sqrtf(na), 1e-12f) * fmaxf(sqrtf(nb), 1e-12f));
  atomicAdd(acc, s);
}

__global__ void k_pur(const float* __restrict__ degs, const float* __restrict__ zq,
                      const float* __restrict__ zk, float* __restrict__ acc_sq,
                      float* __restrict__ acc_cnt) {
  int i = blockIdx.x * blockDim.x + threadIdx.x;
  if (i >= NN) return;
  if (degs[i] != 0.f) return;
  const float* a = zq + (size_t)i * HIDD;
  const float* b = zk + (size_t)i * HIDD;
  float s = 0.f;
  for (int f = 0; f < HIDD; f += 4) {
    fl4 u = *reinterpret_cast<const fl4*>(a + f);
    fl4 v = *reinterpret_cast<const fl4*>(b + f);
    float d0=u[0]-v[0], d1=u[1]-v[1], d2=u[2]-v[2], d3=u[3]-v[3];
    s += d0*d0 + d1*d1 + d2*d2 + d3*d3;
  }
  atomicAdd(acc_sq, s);
  atomicAdd(acc_cnt, 1.f);
}

__global__ void k_cluster(const float* __restrict__ z, const float* __restrict__ centers,
                          float* __restrict__ acc) {
  __shared__ float sc[KC * HIDD];
  __shared__ float cc[KC];
  for (int i = threadIdx.x; i < KC * HIDD; i += blockDim.x) sc[i] = centers[i];
  __syncthreads();
  if (threadIdx.x < KC) {
    float s = 0.f;
    for (int f = 0; f < HIDD; f++) { float c = sc[threadIdx.x * HIDD + f]; s += c * c; }
    cc[threadIdx.x] = s;
  }
  __syncthreads();
  int i = blockIdx.x * blockDim.x + threadIdx.x;
  if (i >= NN) return;
  const float* zr = z + (size_t)i * HIDD;
  float zz = 0.f;
  for (int f = 0; f < HIDD; f += 4) {
    fl4 v = *reinterpret_cast<const fl4*>(zr + f);
    zz += v[0]*v[0] + v[1]*v[1] + v[2]*v[2] + v[3]*v[3];
  }
  float umax = -1.f, usum = 0.f;
  for (int k = 0; k < KC; k++) {
    const float* cr = sc + k * HIDD;
    float dot = 0.f;
    for (int f = 0; f < HIDD; f += 4) {
      fl4 v = *reinterpret_cast<const fl4*>(zr + f);
      dot += v[0]*cr[f] + v[1]*cr[f+1] + v[2]*cr[f+2] + v[3]*cr[f+3];
    }
    float d2 = fmaxf(zz + cc[k] - 2.f * dot, 0.f);
    float u = 1.f / (1.f + d2);
    usum += u;
    if (u > umax) umax = u;
  }
  atomicAdd(acc, logf(umax) - logf(usum));
}

// Mask predictor: gathered WMMA GEMM (16 edges per wave) fused with BCE terms.
__global__ void k_mp_fused(const _Float16* __restrict__ zqh,
                           const int* __restrict__ src, const int* __restrict__ dst,
                           const _Float16* __restrict__ w1t,
                           const float* __restrict__ b1, const float* __restrict__ w2,
                           const float* __restrict__ b2p,
                           const float* __restrict__ wpur, float* __restrict__ acc) {
  const int lane = threadIdx.x;
  const int e0 = blockIdx.x * 16;
  const int m = lane & 15;
  const int kh8 = (lane >> 4) * 8;
  const int e = e0 + m;
  const _Float16* zs = zqh + (size_t)src[e] * HIDD;
  const _Float16* zd = zqh + (size_t)dst[e] * HIDD;
  v8f acc4[4] = {};
  #pragma unroll
  for (int k0 = 0; k0 < 2 * HIDD; k0 += 32) {
    const int kb1 = k0 + kh8;
    const int kb2 = k0 + 16 + kh8;
    half8 lo = *reinterpret_cast<const half8*>((kb1 < HIDD) ? (zs + kb1) : (zd + kb1 - HIDD));
    half8 hi = *reinterpret_cast<const half8*>((kb2 < HIDD) ? (zs + kb2) : (zd + kb2 - HIDD));
    v16h a;
    #pragma unroll
    for (int i = 0; i < 8; i++) { a[i] = lo[i]; a[8 + i] = hi[i]; }
    #pragma unroll
    for (int t = 0; t < 4; t++) {
      v16h b = load_B16(w1t, 2 * HIDD, t * 16, k0, lane);
      acc4[t] = wmma16(a, b, acc4[t]);
    }
  }
  const int col = lane & 15;
  const int rb  = (lane >> 4) * 8;
  float s[8];
  #pragma unroll
  for (int i = 0; i < 8; i++) s[i] = 0.f;
  #pragma unroll
  for (int t = 0; t < 4; t++) {
    const float bv = b1[t * 16 + col];
    const float wv = w2[t * 16 + col];
    #pragma unroll
    for (int i = 0; i < 8; i++) {
      float h = fmaxf(acc4[t][i] + bv, 0.f);
      s[i] += h * wv;
    }
  }
  #pragma unroll
  for (int mm = 1; mm <= 8; mm <<= 1)
    #pragma unroll
    for (int i = 0; i < 8; i++) s[i] += __shfl_xor(s[i], mm, 32);
  if ((lane & 15) == 0) {
    const float b2 = b2p[0];
    float lp = 0.f, ln = 0.f, nm = 0.f;
    #pragma unroll
    for (int i = 0; i < 8; i++) {
      int ei = e0 + rb + i;
      float pred = 1.f / (1.f + __expf(-(s[i] + b2)));
      float wm = 1.f - wpur[ei];
      lp += logf(pred + 1e-15f);
      ln += wm * logf(1.f - pred + 1e-15f);
      nm += wm;
    }
    atomicAdd(&acc[3], lp);
    atomicAdd(&acc[4], ln);
    atomicAdd(&acc[5], nm);
  }
}

__global__ void k_final(const float* __restrict__ acc, float* __restrict__ out) {
  if (threadIdx.x != 0 || blockIdx.x != 0) return;
  const float Ef = (float)EE, Nf = (float)NN;
  float l1 = -(acc[0] / Ef) - (acc[1] / Ef);
  float l2 = acc[2] / (Nf * (float)FDIM);
  float nm = acc[5];
  float w_pos = (nm + Ef) / (2.f * Ef);
  float w_neg = (nm + Ef) / (2.f * fmaxf(nm, 1.f));
  float lp = acc[3] / Ef;
  float ln = acc[4] / fmaxf(nm, 1.f);
  float l_ban = (nm > 0.f) ? (-w_pos * lp - w_neg * ln) : 0.f;
  float l_fg = -(acc[6] / Nf);
  float l_pur = (acc[8] > 0.f) ? (acc[7] / fmaxf(acc[8] * (float)HIDD, 1.f)) : 0.f;
  float l_clu = -(acc[9] / Nf);
  out[0] = l1 + l2 + l_ban + 0.1f * l_fg + 0.01f * l_pur + 0.001f * l_clu;
}

// ---------------------------------------------------------------------------
// Launch
// ---------------------------------------------------------------------------
extern "C" void kernel_launch(void* const* d_in, const int* in_sizes, int n_in,
                              void* d_out, int out_size, void* d_ws, size_t ws_size,
                              hipStream_t stream) {
  (void)in_sizes; (void)n_in; (void)out_size; (void)ws_size;
  const float* x      = (const float*)d_in[0];
  const float* mnoise = (const float*)d_in[1];
  const float* perm   = (const float*)d_in[2];
  const int*   ei     = (const int*)d_in[3];
  const int*   nei    = (const int*)d_in[4];
  const float* enc_w1 = (const float*)d_in[5];  const float* enc_b1 = (const float*)d_in[6];
  const float* enc_w2 = (const float*)d_in[7];  const float* enc_b2 = (const float*)d_in[8];
  const float* g1w1   = (const float*)d_in[9];  const float* g1b1   = (const float*)d_in[10];
  const float* g1w2   = (const float*)d_in[11]; const float* g1b2   = (const float*)d_in[12];
  const float* g2w1   = (const float*)d_in[13]; const float* g2b1   = (const float*)d_in[14];
  const float* g2w2   = (const float*)d_in[15]; const float* g2b2   = (const float*)d_in[16];
  const float* fdw1   = (const float*)d_in[17]; const float* fdb1   = (const float*)d_in[18];
  const float* fdw2   = (const float*)d_in[19]; const float* fdb2   = (const float*)d_in[20];
  const float* mpw1   = (const float*)d_in[21]; const float* mpb1   = (const float*)d_in[22];
  const float* mpw2   = (const float*)d_in[23]; const float* mpb2   = (const float*)d_in[24];
  const float* centers= (const float*)d_in[25];
  const int* src = ei;
  const int* dst = ei + EE;

  // workspace carve-out (256B aligned)
  char* base = (char*)d_ws;
  size_t off = 0;
  auto alloc = [&](size_t bytes) -> void* {
    void* p = base + off;
    off = (off + bytes + 255) & ~(size_t)255;
    return p;
  };
  float*     ACC   = (float*)alloc(32 * 4);
  unsigned*  CTRL  = (unsigned*)alloc(32 * 4);
  unsigned*  HIST  = (unsigned*)alloc(256 * 4);
  float*     NORM  = (float*)alloc((size_t)NN * 4);
  float*     SIM   = (float*)alloc((size_t)EE * 4);
  unsigned*  KEYS  = (unsigned*)alloc((size_t)EE * 4);
  float*     WPUR  = (float*)alloc((size_t)EE * 4);
  float*     WV2   = (float*)alloc((size_t)EE * 4);
  float*     DEGQ  = (float*)alloc((size_t)NN * 4);
  float*     DEGK  = (float*)alloc((size_t)NN * 4);
  float*     DEGS  = (float*)alloc((size_t)NN * 4);
  _Float16*  XHT   = (_Float16*)alloc((size_t)NN * FDIM * 2);
  _Float16*  W1T   = (_Float16*)alloc((size_t)FDIM * ENCD * 2);
  _Float16*  W2T   = (_Float16*)alloc((size_t)ENCD * HIDD * 2);
  _Float16*  G1W1T = (_Float16*)alloc((size_t)HIDD * DECD * 2);
  _Float16*  G1W2T = (_Float16*)alloc((size_t)DECD * DECD * 2);
  _Float16*  G2W1T = (_Float16*)alloc((size_t)DECD * DECD * 2);
  _Float16*  G2W2T = (_Float16*)alloc((size_t)DECD * DECD * 2);
  _Float16*  FDW1T = (_Float16*)alloc((size_t)HIDD * DECD * 2);
  _Float16*  FDW2T = (_Float16*)alloc((size_t)DECD * FDIM * 2);
  _Float16*  MPW1T = (_Float16*)alloc((size_t)2 * HIDD * DECD * 2);
  _Float16*  XBANH = (_Float16*)alloc((size_t)NN * FDIM * 2);
  float*     A1    = (float*)alloc((size_t)NN * ENCD * 4);
  float*     A2    = (float*)alloc((size_t)NN * ENCD * 4);
  _Float16*  B1H   = (_Float16*)alloc((size_t)NN * ENCD * 2);
  _Float16*  B2H   = (_Float16*)alloc((size_t)NN * ENCD * 2);
  float*     A3    = (float*)alloc((size_t)NN * HIDD * 4);
  float*     A4    = (float*)alloc((size_t)NN * HIDD * 4);
  float*     Z1    = (float*)alloc((size_t)NN * HIDD * 4);
  float*     Z2    = (float*)alloc((size_t)NN * HIDD * 4);
  _Float16*  Z1H   = (_Float16*)alloc((size_t)NN * HIDD * 2);
  _Float16*  GH1   = (_Float16*)alloc((size_t)NN * HIDD * 2);
  _Float16*  GH2   = (_Float16*)alloc((size_t)NN * DECD * 2);
  _Float16*  GH3   = (_Float16*)alloc((size_t)NN * DECD * 2);
  float*     G1B   = (float*)alloc((size_t)NN * DECD * 4);
  float*     H1    = (float*)alloc((size_t)NN * DECD * 4);
  float*     H2    = (float*)alloc((size_t)NN * DECD * 4);

  const int EB = EE / 256;           // per-edge kernels
  const int NB = NN / 256;           // per-node kernels

  // --- init accumulators / control ---
  k_fill_f32<<<1, 32, 0, stream>>>(ACC, 32, 0.f);
  k_zero_u32<<<1, 32, 0, stream>>>(CTRL, 32);
  k_zero_u32<<<1, 256, 0, stream>>>(HIST, 256);

  // --- weight / input conversion (f32 -> f16, transpose to K-contiguous) ---
  k_transpose_h<<<(NN * FDIM) / 256, 256, 0, stream>>>(x, XHT, NN, FDIM);
  k_transpose_h<<<(FDIM * ENCD) / 256, 256, 0, stream>>>(enc_w1, W1T, FDIM, ENCD);
  k_transpose_h<<<(ENCD * HIDD) / 256, 256, 0, stream>>>(enc_w2, W2T, ENCD, HIDD);
  k_transpose_h<<<(HIDD * DECD) / 256, 256, 0, stream>>>(g1w1, G1W1T, HIDD, DECD);
  k_transpose_h<<<(DECD * DECD) / 256, 256, 0, stream>>>(g1w2, G1W2T, DECD, DECD);
  k_transpose_h<<<(DECD * DECD) / 256, 256, 0, stream>>>(g2w1, G2W1T, DECD, DECD);
  k_transpose_h<<<(DECD * DECD) / 256, 256, 0, stream>>>(g2w2, G2W2T, DECD, DECD);
  k_transpose_h<<<(HIDD * DECD) / 256, 256, 0, stream>>>(fdw1, FDW1T, HIDD, DECD);
  k_transpose_h<<<(DECD * FDIM) / 256, 256, 0, stream>>>(fdw2, FDW2T, DECD, FDIM);
  k_transpose_h<<<(2 * HIDD * DECD) / 256, 256, 0, stream>>>(mpw1, MPW1T, 2 * HIDD, DECD);

  // --- x_ban = softmax(mask/alpha) @ x  (single pass, online softmax, WMMA) ---
  k_softmax_gemm<<<NN / 64, 128, 0, stream>>>(mnoise, XHT, XBANH);

  // --- topology filter + recovery top-k ---
  k_norms<<<NB, 256, 0, stream>>>(x, NORM);
  k_sim<<<EB, 256, 0, stream>>>(x, NORM, src, dst, SIM, KEYS, CTRL);
  k_sel_init<<<1, 1, 0, stream>>>(CTRL, 0);
  {
    const unsigned masks[4] = {0x00000000u, 0xFF000000u, 0xFFFF0000u, 0xFFFFFF00u};
    const int shifts[4] = {24, 16, 8, 0};
    for (int p = 0; p < 4; p++) {
      k_hist<<<EB, 256, 0, stream>>>(KEYS, CTRL, HIST, masks[p], shifts[p]);
      k_sel_bin<<<1, 1, 0, stream>>>(CTRL, HIST, shifts[p]);
    }
  }
  k_mark_pur<<<EB, 256, 0, stream>>>(SIM, KEYS, CTRL, WPUR);

  // --- 70% random sub-view of purified edges ---
  k_keys_v2<<<EB, 256, 0, stream>>>(perm, WPUR, KEYS);
  k_sel_init<<<1, 1, 0, stream>>>(CTRL, 1);
  {
    const unsigned masks[4] = {0x00000000u, 0xFF000000u, 0xFFFF0000u, 0xFFFFFF00u};
    const int shifts[4] = {24, 16, 8, 0};
    for (int p = 0; p < 4; p++) {
      k_hist<<<EB, 256, 0, stream>>>(KEYS, CTRL, HIST, masks[p], shifts[p]);
      k_sel_bin<<<1, 1, 0, stream>>>(CTRL, HIST, shifts[p]);
    }
  }
  k_mark_v2<<<EB, 256, 0, stream>>>(KEYS, WPUR, CTRL, WV2);

  // --- degrees ---
  k_fill_f32<<<NB, 256, 0, stream>>>(DEGQ, NN, 1.f);
  k_deg_add<<<EB, 256, 0, stream>>>(dst, (const float*)nullptr, DEGQ);
  k_fill_f32<<<NB, 256, 0, stream>>>(DEGK, NN, 1.f);
  k_deg_add<<<EB, 256, 0, stream>>>(dst, WPUR, DEGK);
  k_fill_f32<<<NB, 256, 0, stream>>>(DEGS, NN, 0.f);
  k_deg_add<<<EB, 256, 0, stream>>>(src, WPUR, DEGS);

  const int SCB = (EE * 32) / 256;   // scatter blocks
  // --- encoder layer 1 (shared hl) ---
  k_gemm<<<dim3(NN / 64, ENCD / 64), 128, 0, stream>>>(XBANH, FDIM, W1T, FDIM,
      nullptr, A1, nullptr, ENCD, FDIM, 0);
  // q branch
  k_fill_f32<<<4096, 256, 0, stream>>>(A2, (long long)NN * ENCD, 0.f);
  k_scatter<<<SCB, 256, 0, stream>>>(A1, ENCD, src, dst, (const float*)nullptr, DEGQ, A2);
  k_gcn_combine<<<(NN * ENCD) / 256, 256, 0, stream>>>(A2, A1, DEGQ, enc_b1, ENCD, 1, nullptr, B1H);
  // k branch
  k_fill_f32<<<4096, 256, 0, stream>>>(A2, (long long)NN * ENCD, 0.f);
  k_scatter<<<SCB, 256, 0, stream>>>(A1, ENCD, src, dst, WPUR, DEGK, A2);
  k_gcn_combine<<<(NN * ENCD) / 256, 256, 0, stream>>>(A2, A1, DEGK, enc_b1, ENCD, 1, nullptr, B2H);

  // --- encoder layer 2: z_q ---
  k_gemm<<<dim3(NN / 64, HIDD / 64), 128, 0, stream>>>(B1H, ENCD, W2T, ENCD,
      nullptr, A3, nullptr, HIDD, ENCD, 0);
  k_fill_f32<<<4096, 256, 0, stream>>>(A4, (long long)NN * HIDD, 0.f);
  k_scatter<<<SCB, 256, 0, stream>>>(A3, HIDD, src, dst, (const float*)nullptr, DEGQ, A4);
  k_gcn_combine<<<(NN * HIDD) / 256, 256, 0, stream>>>(A4, A3, DEGQ, enc_b2, HIDD, 0, Z1, Z1H);
  // --- encoder layer 2: z_k ---
  k_gemm<<<dim3(NN / 64, HIDD / 64), 128, 0, stream>>>(B2H, ENCD, W2T, ENCD,
      nullptr, A3, nullptr, HIDD, ENCD, 0);
  k_fill_f32<<<4096, 256, 0, stream>>>(A4, (long long)NN * HIDD, 0.f);
  k_scatter<<<SCB, 256, 0, stream>>>(A3, HIDD, src, dst, WPUR, DEGK, A4);
  k_gcn_combine<<<(NN * HIDD) / 256, 256, 0, stream>>>(A4, A3, DEGK, enc_b2, HIDD, 0, Z2, nullptr);

  // --- GIN runs (on z_k) ---
  auto gin_run = [&](const float* W, float* Hout) {
    // layer 1
    k_fill_f32<<<4096, 256, 0, stream>>>(A4, (long long)NN * HIDD, 0.f);
    k_scatter<<<SCB, 256, 0, stream>>>(Z2, HIDD, src, dst, W, (const float*)nullptr, A4);
    k_gin_combine<<<(NN * HIDD) / 256, 256, 0, stream>>>(A4, Z2, GH1, (long long)NN * HIDD);
    k_gemm<<<dim3(NN / 64, 1), 128, 0, stream>>>(GH1, HIDD, G1W1T, HIDD, g1b1,
        nullptr, GH2, DECD, HIDD, 1);
    k_gemm<<<dim3(NN / 64, 1), 128, 0, stream>>>(GH2, DECD, G1W2T, DECD, g1b2,
        G1B, nullptr, DECD, DECD, 0);
    // layer 2
    k_fill_f32<<<2048, 256, 0, stream>>>(A4, (long long)NN * DECD, 0.f);
    k_scatter<<<SCB, 256, 0, stream>>>(G1B, DECD, src, dst, W, (const float*)nullptr, A4);
    k_gin_combine<<<(NN * DECD) / 256, 256, 0, stream>>>(A4, G1B, GH2, (long long)NN * DECD);
    k_gemm<<<dim3(NN / 64, 1), 128, 0, stream>>>(GH2, DECD, G2W1T, DECD, g2b1,
        nullptr, GH3, DECD, DECD, 1);
    k_gemm<<<dim3(NN / 64, 1), 128, 0, stream>>>(GH3, DECD, G2W2T, DECD, g2b2,
        Hout, nullptr, DECD, DECD, 0);
  };
  gin_run(WPUR, H1);
  gin_run(WV2, H2);

  // --- losses ---
  k_fg<<<NB, 256, 0, stream>>>(H1, H2, &ACC[6]);
  k_pur<<<NB, 256, 0, stream>>>(DEGS, Z1, Z2, &ACC[7], &ACC[8]);
  k_cluster<<<NN / 128, 128, 0, stream>>>(Z1, centers, &ACC[9]);
  k_linkloss<<<EB, 256, 0, stream>>>(Z1, ei, 1.f, &ACC[0]);
  k_linkloss<<<EB, 256, 0, stream>>>(Z1, nei, -1.f, &ACC[1]);

  // feature decoder + MSE
  k_gemm<<<dim3(NN / 64, 1), 128, 0, stream>>>(Z1H, HIDD, FDW1T, HIDD, fdb1,
      nullptr, GH2, DECD, HIDD, 1);
  k_gemm<<<dim3(NN / 64, FDIM / 64), 128, 0, stream>>>(GH2, DECD, FDW2T, DECD, fdb2,
      A3, nullptr, FDIM, DECD, 0);
  k_mse<<<256, 256, 0, stream>>>(A3, x, (long long)NN * FDIM, &ACC[2]);

  // mask predictor (gathered WMMA GEMM + fused BCE)
  k_mp_fused<<<EE / 16, 32, 0, stream>>>(Z1H, src, dst, MPW1T, mpb1, mpw2, mpb2,
      WPUR, ACC);

  // --- final combine ---
  k_final<<<1, 1, 0, stream>>>(ACC, (float*)d_out);
}
